// PropNet_25632364822768
// MI455X (gfx1250) — matrix operand order repaired and checked by reading, not compiled, not run on hardware
//
#include <hip/hip_runtime.h>

// ---------------------------------------------------------------------------
// CDNA5 (gfx1250) PropNet: bf16 WMMA (V_WMMA_F32_16X16X32_BF16) pipeline,
// f32 accumulation, wave32. Register-blocked: one wave owns a 32x64 output
// block (2 m-tiles x 4 n-tiles) -> 8 WMMAs per 12 b128 loads per K-step.
// ---------------------------------------------------------------------------

typedef __bf16 bf16;
typedef __attribute__((ext_vector_type(8)))  __bf16 v8bf;
typedef __attribute__((ext_vector_type(16))) __bf16 v16bf;
typedef __attribute__((ext_vector_type(8)))  float  v8f;

#define BBATCH 2
#define NN     64
#define RR     16
#define DNODE  64
#define HDIM   256

// --- A fragment: 16x32 bf16 tile (ISA 7.12.2). Lane L (lr=L&15, half=L>>4)
// holds row m=lr; lanes 0-15: K = k0+0..7 and k0+16..23; lanes 16-31:
// K = k0+8..15 and k0+24..31. Two contiguous 16B chunks per lane.
static __device__ __forceinline__ v16bf frag_a(const bf16* __restrict__ A, int ld,
                                               long m0, int k0, int lr, int half) {
  const bf16* p = A + (m0 + lr) * (long)ld + (k0 + half * 8);
  v8bf lo = *(const v8bf*)p;
  v8bf hi = *(const v8bf*)(p + 16);
  return __builtin_shufflevector(lo, hi, 0,1,2,3,4,5,6,7,8,9,10,11,12,13,14,15);
}

// --- B fragment: 32x16 bf16 tile. Lane L holds column n=lr; lanes 0-15 hold
// K=k0+0..7 and k0+16..23, lanes 16-31 hold K=k0+8..15 and k0+24..31. Loaded
// from PRE-TRANSPOSED weight Wt[N][K]: two contiguous 16B chunks per lane.
static __device__ __forceinline__ v16bf frag_b(const bf16* __restrict__ Wt, int ld,
                                               int n0, int k0, int lr, int half) {
  const bf16* p = Wt + (long)(n0 + lr) * ld + (k0 + half * 8);
  v8bf lo = *(const v8bf*)p;
  v8bf hi = *(const v8bf*)(p + 16);
  return __builtin_shufflevector(lo, hi, 0,1,2,3,4,5,6,7,8,9,10,11,12,13,14,15);
}

static __device__ __forceinline__ v8f wmma_bf16(v16bf a, v16bf b, v8f c) {
  // 8-arg form: (neg_a, A, neg_b, B, c_mod, C, reuse_a, reuse_b)
  return __builtin_amdgcn_wmma_f32_16x16x32_bf16(false, a, false, b, (short)0, c,
                                                 false, false);
}

// --------------------------- prep kernels ----------------------------------
__global__ void k_tcvt(const float* __restrict__ W, bf16* __restrict__ Wt,
                       int K, int N) {
  int idx = blockIdx.x * blockDim.x + threadIdx.x;
  if (idx >= K * N) return;
  int n = idx / K, k = idx % K;
  Wt[idx] = (bf16)W[(long)k * N + n];
}

__global__ void k_cvt(const float* __restrict__ x, bf16* __restrict__ y, int n) {
  int i = blockIdx.x * blockDim.x + threadIdx.x;
  if (i < n) y[i] = (bf16)x[i];
}

// --------------------------- generic GEMM ----------------------------------
// D = sum_p A_p @ W_p [+bias] [relu], up to 3 pairs sharing K.
// One wave per 32x64 block. Requires M%32==0, N%64==0.
__global__ void k_gemm(const bf16* __restrict__ A0, const bf16* __restrict__ W0,
                       const bf16* __restrict__ A1, const bf16* __restrict__ W1,
                       const bf16* __restrict__ A2, const bf16* __restrict__ W2,
                       const float* __restrict__ bias,
                       long M, int K, int N, int do_relu,
                       float* __restrict__ outF, bf16* __restrict__ outB) {
  int wid  = (blockIdx.x * blockDim.x + threadIdx.x) >> 5;
  int lane = threadIdx.x & 31;
  int nblk = N >> 6;
  long jobs = (M >> 5) * nblk;
  if (wid >= jobs) return;                    // wave-uniform: EXEC stays full
  long mt = wid / nblk;
  int  nt = wid % nblk;
  long m0 = mt << 5;
  int  n0 = nt << 6;
  int  lr = lane & 15, half = lane >> 4;

  v8f acc[2][4] = {};
  auto mma_pair = [&](const bf16* __restrict__ A, const bf16* __restrict__ W) {
    for (int k0 = 0; k0 < K; k0 += 32) {
      v16bf a0 = frag_a(A, K, m0,      k0, lr, half);
      v16bf a1 = frag_a(A, K, m0 + 16, k0, lr, half);
#pragma unroll
      for (int ni = 0; ni < 4; ++ni) {
        v16bf bb = frag_b(W, K, n0 + ni * 16, k0, lr, half);
        acc[0][ni] = wmma_bf16(a0, bb, acc[0][ni]);
        acc[1][ni] = wmma_bf16(a1, bb, acc[1][ni]);
      }
    }
  };
  if (W0) mma_pair(A0, W0);
  if (W1) mma_pair(A1, W1);
  if (W2) mma_pair(A2, W2);

#pragma unroll
  for (int mi = 0; mi < 2; ++mi) {
#pragma unroll
    for (int ni = 0; ni < 4; ++ni) {
      int n = n0 + ni * 16 + lr;
      float bv = bias ? bias[n] : 0.f;
#pragma unroll
      for (int v = 0; v < 8; ++v) {
        int m = half * 8 + v;                 // D layout: VGPR v -> row v(+8)
        float d = acc[mi][ni][v] + bv;
        if (do_relu) d = d > 0.f ? d : 0.f;
        long o = (m0 + mi * 16 + m) * (long)N + n;
        if (outF) outF[o] = d;
        if (outB) outB[o] = (bf16)d;
      }
    }
  }
}

// --------------------------- big edge GEMM #1 ------------------------------
// h[e] = relu( [Estored @ WtE] + [edge_enc(virtual) @ WtV] + Er[b,i,r] +
//              Es[b,j,r] + bias ),  M = B*N*N*R = 131072, K = 256.
// edge_enc is synthesized per A-fragment as relu(Xr[b,i,r]+Xs[b,j,r]+eeb)
// (never materialized). A 32-row block = fixed (b,i), senders (j, j+1).
__global__ void k_edge1(const bf16* __restrict__ Ein, const bf16* __restrict__ WtE,
                        const float* __restrict__ Xr, const float* __restrict__ Xs,
                        const float* __restrict__ eeb, const bf16* __restrict__ WtV,
                        const float* __restrict__ Er, const float* __restrict__ Es,
                        const float* __restrict__ bias, bf16* __restrict__ out) {
  const int nblk = HDIM >> 6;                 // 4
  int wid  = (blockIdx.x * blockDim.x + threadIdx.x) >> 5;
  int lane = threadIdx.x & 31;
  int jobs = ((BBATCH * NN * NN * RR) >> 5) * nblk;   // 16384 waves
  if (wid >= jobs) return;
  int mt = wid / nblk, nt = wid % nblk;
  long e0 = (long)mt << 5;                    // 32 rows: (j,r=0..15),(j+1,r=0..15)
  int  n0 = nt << 6;
  int  lr = lane & 15, half = lane >> 4;

  int j = (int)((e0 >> 4) & (NN - 1));        // even
  int i = (int)((e0 >> 10) & (NN - 1));
  int b = (int)(e0 >> 16);
  long ri = (long)(b * NN + i) * RR;          // receiver node-row base (r=0)
  long rj = (long)(b * NN + j) * RR;          // sender base (tile1: rj+16)

  v8f acc[2][4] = {};
  if (WtE) {
    for (int k0 = 0; k0 < HDIM; k0 += 32) {
      v16bf a0 = frag_a(Ein, HDIM, e0,      k0, lr, half);
      v16bf a1 = frag_a(Ein, HDIM, e0 + 16, k0, lr, half);
#pragma unroll
      for (int ni = 0; ni < 4; ++ni) {
        v16bf bb = frag_b(WtE, HDIM, n0 + ni * 16, k0, lr, half);
        acc[0][ni] = wmma_bf16(a0, bb, acc[0][ni]);
        acc[1][ni] = wmma_bf16(a1, bb, acc[1][ni]);
      }
    }
  }
  if (WtV) {
    const float* xr  = Xr + (ri + lr) * HDIM;       // shared receiver term
    const float* xs0 = Xs + (rj + lr) * HDIM;       // sender j
    const float* xs1 = Xs + (rj + 16 + lr) * HDIM;  // sender j+1
    for (int k0 = 0; k0 < HDIM; k0 += 32) {
      int kb = k0 + half * 8;
      v16bf a0, a1;
#pragma unroll
      for (int t = 0; t < 8; ++t) {
        float c_lo = xr[kb + t]      + eeb[kb + t];
        float c_hi = xr[kb + t + 16] + eeb[kb + t + 16];
        float l0 = c_lo + xs0[kb + t],      h0 = c_hi + xs0[kb + t + 16];
        float l1 = c_lo + xs1[kb + t],      h1 = c_hi + xs1[kb + t + 16];
        a0[t] = (bf16)(l0 > 0.f ? l0 : 0.f);  a0[t + 8] = (bf16)(h0 > 0.f ? h0 : 0.f);
        a1[t] = (bf16)(l1 > 0.f ? l1 : 0.f);  a1[t + 8] = (bf16)(h1 > 0.f ? h1 : 0.f);
      }
#pragma unroll
      for (int ni = 0; ni < 4; ++ni) {
        v16bf bb = frag_b(WtV, HDIM, n0 + ni * 16, k0, lr, half);
        acc[0][ni] = wmma_bf16(a0, bb, acc[0][ni]);
        acc[1][ni] = wmma_bf16(a1, bb, acc[1][ni]);
      }
    }
  }
#pragma unroll
  for (int mi = 0; mi < 2; ++mi) {
    long eb  = e0 + mi * 16;
    long rjm = rj + mi * 16;
#pragma unroll
    for (int ni = 0; ni < 4; ++ni) {
      int n = n0 + ni * 16 + lr;
      float bv = bias[n];
#pragma unroll
      for (int v = 0; v < 8; ++v) {
        int m = half * 8 + v;                 // row m == r within the tile
        float d = acc[mi][ni][v] + bv;
        if (Er) d += Er[(ri + m) * HDIM + n] + Es[(rjm + m) * HDIM + n];
        d = d > 0.f ? d : 0.f;
        out[(eb + m) * HDIM + n] = (bf16)d;
      }
    }
  }
}

// --------------------------- big edge GEMM #2 ------------------------------
// Enew = relu(h @ W2 + b2), fused sender-sum: one wave owns (b,i,n-group),
// loops sender pairs, accumulates agg[b,i,r,n] in registers (no atomics).
__global__ void k_edge2(const bf16* __restrict__ Hin, const bf16* __restrict__ W2t,
                        const float* __restrict__ b2,
                        bf16* __restrict__ Eout, float* __restrict__ agg) {
  const int nblk = HDIM >> 6;                 // 4
  int wid  = (blockIdx.x * blockDim.x + threadIdx.x) >> 5;
  int lane = threadIdx.x & 31;
  int jobs = BBATCH * NN * nblk;              // 512 waves
  if (wid >= jobs) return;
  int bi = wid / nblk, nt = wid % nblk;
  int n0 = nt << 6;
  int lr = lane & 15, half = lane >> 4;

  float bv[4];
#pragma unroll
  for (int ni = 0; ni < 4; ++ni) bv[ni] = b2[n0 + ni * 16 + lr];

  v8f aggv[4] = {};
  for (int j0 = 0; j0 < NN; j0 += 2) {
    long e0 = ((long)bi * NN + j0) * RR;      // 32 rows: senders j0, j0+1
    v8f acc[2][4] = {};
    for (int k0 = 0; k0 < HDIM; k0 += 32) {
      v16bf a0 = frag_a(Hin, HDIM, e0,      k0, lr, half);
      v16bf a1 = frag_a(Hin, HDIM, e0 + 16, k0, lr, half);
#pragma unroll
      for (int ni = 0; ni < 4; ++ni) {
        v16bf bb = frag_b(W2t, HDIM, n0 + ni * 16, k0, lr, half);
        acc[0][ni] = wmma_bf16(a0, bb, acc[0][ni]);
        acc[1][ni] = wmma_bf16(a1, bb, acc[1][ni]);
      }
    }
#pragma unroll
    for (int mi = 0; mi < 2; ++mi) {
#pragma unroll
      for (int ni = 0; ni < 4; ++ni) {
        int n = n0 + ni * 16 + lr;
#pragma unroll
        for (int v = 0; v < 8; ++v) {
          int m = half * 8 + v;
          float d = acc[mi][ni][v] + bv[ni];
          d = d > 0.f ? d : 0.f;
          Eout[(e0 + mi * 16 + m) * HDIM + n] = (bf16)d;
          aggv[ni][v] += d;                   // both senders -> same agg row r=m
        }
      }
    }
  }
#pragma unroll
  for (int ni = 0; ni < 4; ++ni) {
#pragma unroll
    for (int v = 0; v < 8; ++v) {
      int m = half * 8 + v;
      agg[((long)bi * RR + m) * HDIM + n0 + ni * 16 + lr] = aggv[ni][v];
    }
  }
}

// ---------------------------------------------------------------------------
extern "C" void kernel_launch(void* const* d_in, const int* in_sizes, int n_in,
                              void* d_out, int out_size, void* d_ws, size_t ws_size,
                              hipStream_t stream) {
  (void)in_sizes; (void)n_in; (void)out_size; (void)ws_size;
  const float* node_rep = (const float*)d_in[0];
  const float* ne_W = (const float*)d_in[1];  const float* ne_b = (const float*)d_in[2];
  const float* ee_W = (const float*)d_in[3];  const float* ee_b = (const float*)d_in[4];
  const float* np_W1 = (const float*)d_in[5]; const float* np_b1 = (const float*)d_in[6];
  const float* np_W2 = (const float*)d_in[7]; const float* np_b2 = (const float*)d_in[8];
  const float* epW1[2] = {(const float*)d_in[9],  (const float*)d_in[13]};
  const float* epb1[2] = {(const float*)d_in[10], (const float*)d_in[14]};
  const float* epW2[2] = {(const float*)d_in[11], (const float*)d_in[15]};
  const float* epb2[2] = {(const float*)d_in[12], (const float*)d_in[16]};
  const float* pr_W1 = (const float*)d_in[17]; const float* pr_b1 = (const float*)d_in[18];
  const float* pr_W2 = (const float*)d_in[19]; const float* pr_b2 = (const float*)d_in[20];

  char* ws = (char*)d_ws;  size_t off = 0;
  auto alloc = [&](size_t bytes) -> void* {
    void* p = ws + off; off += (bytes + 255) & ~(size_t)255; return p;
  };
  const long NRows = (long)BBATCH * NN * RR;        // 2048
  const long ERows = (long)BBATCH * NN * NN * RR;   // 131072

  // bf16 transposed weights
  bf16* neT   = (bf16*)alloc((size_t)DNODE * HDIM * 2);
  bf16* eeAT  = (bf16*)alloc((size_t)DNODE * HDIM * 2);
  bf16* eeBT  = (bf16*)alloc((size_t)DNODE * HDIM * 2);
  bf16* npAT  = (bf16*)alloc((size_t)HDIM * HDIM * 2);
  bf16* npBT  = (bf16*)alloc((size_t)HDIM * HDIM * 2);
  bf16* npCT  = (bf16*)alloc((size_t)HDIM * HDIM * 2);
  bf16* npW2T = (bf16*)alloc((size_t)HDIM * HDIM * 2);
  bf16* epAT[2], *epBT[2], *epCT[2], *epW2T[2];
  for (int t = 0; t < 2; ++t) {
    epAT[t]  = (bf16*)alloc((size_t)HDIM * HDIM * 2);
    epBT[t]  = (bf16*)alloc((size_t)HDIM * HDIM * 2);
    epCT[t]  = (bf16*)alloc((size_t)HDIM * HDIM * 2);
    epW2T[t] = (bf16*)alloc((size_t)HDIM * HDIM * 2);
  }
  bf16* prAT  = (bf16*)alloc((size_t)HDIM * HDIM * 2);
  bf16* prBT  = (bf16*)alloc((size_t)HDIM * HDIM * 2);
  bf16* prW2T = (bf16*)alloc((size_t)HDIM * 64 * 2);
  // node-level buffers
  bf16*  nrB      = (bf16*)alloc((size_t)NRows * DNODE * 2);
  bf16*  node_enc = (bf16*)alloc((size_t)NRows * HDIM * 2);
  bf16*  node_eff = (bf16*)alloc((size_t)NRows * HDIM * 2);
  bf16*  gbuf     = (bf16*)alloc((size_t)NRows * HDIM * 2);
  bf16*  aggB     = (bf16*)alloc((size_t)NRows * HDIM * 2);
  float* Xr  = (float*)alloc((size_t)NRows * HDIM * 4);
  float* Xs  = (float*)alloc((size_t)NRows * HDIM * 4);
  float* Er  = (float*)alloc((size_t)NRows * HDIM * 4);
  float* Es  = (float*)alloc((size_t)NRows * HDIM * 4);
  float* agg = (float*)alloc((size_t)NRows * HDIM * 4);
  // big edge buffers (h reused as predictor hidden p)
  bf16* hbuf = (bf16*)alloc((size_t)ERows * HDIM * 2);
  bf16* ebuf = (bf16*)alloc((size_t)ERows * HDIM * 2);

  const int TPB = 256;
  auto blocks1d = [](long n, int tpb) { return dim3((unsigned)((n + tpb - 1) / tpb)); };
  auto tcvt = [&](const float* src, bf16* dst, int K, int N) {
    k_tcvt<<<blocks1d((long)K * N, TPB), TPB, 0, stream>>>(src, dst, K, N);
  };
  auto gemm = [&](const bf16* A0, const bf16* W0, const bf16* A1, const bf16* W1,
                  const bf16* A2, const bf16* W2, const float* bias,
                  long M, int K, int N, int relu, float* oF, bf16* oB) {
    long waves = (M >> 5) * (N >> 6);
    k_gemm<<<blocks1d(waves * 32, TPB), TPB, 0, stream>>>(
        A0, W0, A1, W1, A2, W2, bias, M, K, N, relu, oF, oB);
  };

  // --- weight prep (weights total ~2MB bf16; L2-resident afterwards) ---
  tcvt(ne_W, neT, DNODE, HDIM);
  tcvt(ee_W,               eeAT, DNODE, HDIM);
  tcvt(ee_W + DNODE*HDIM,  eeBT, DNODE, HDIM);
  tcvt(np_W1,               npAT, HDIM, HDIM);
  tcvt(np_W1 + HDIM*HDIM,   npBT, HDIM, HDIM);
  tcvt(np_W1 + 2*HDIM*HDIM, npCT, HDIM, HDIM);
  tcvt(np_W2, npW2T, HDIM, HDIM);
  for (int t = 0; t < 2; ++t) {
    tcvt(epW1[t],               epAT[t], HDIM, HDIM);
    tcvt(epW1[t] + HDIM*HDIM,   epBT[t], HDIM, HDIM);
    tcvt(epW1[t] + 2*HDIM*HDIM, epCT[t], HDIM, HDIM);
    tcvt(epW2[t], epW2T[t], HDIM, HDIM);
  }
  tcvt(pr_W1,             prAT, HDIM, HDIM);
  tcvt(pr_W1 + HDIM*HDIM, prBT, HDIM, HDIM);
  tcvt(pr_W2, prW2T, HDIM, 64);
  k_cvt<<<blocks1d(NRows * DNODE, TPB), TPB, 0, stream>>>(node_rep, nrB,
                                                          (int)(NRows * DNODE));

  // --- encoders ---
  gemm(nrB, neT,  0,0, 0,0, ne_b,   NRows, DNODE, HDIM, 1, nullptr, node_enc);
  gemm(nrB, eeAT, 0,0, 0,0, nullptr, NRows, DNODE, HDIM, 0, Xr, nullptr);
  gemm(nrB, eeBT, 0,0, 0,0, nullptr, NRows, DNODE, HDIM, 0, Xs, nullptr);

  // --- 2 propagation steps ---
  long ewaves = (ERows >> 5) * (HDIM >> 6);   // 16384 waves
  for (int t = 0; t < 2; ++t) {
    const bf16* neff = (t == 0) ? node_enc : node_eff;
    gemm(neff, epAT[t], 0,0, 0,0, nullptr, NRows, HDIM, HDIM, 0, Er, nullptr);
    gemm(neff, epBT[t], 0,0, 0,0, nullptr, NRows, HDIM, HDIM, 0, Es, nullptr);
    if (t == 0) {       // edge input = virtual edge_enc
      k_edge1<<<blocks1d(ewaves * 32, TPB), TPB, 0, stream>>>(
          nullptr, nullptr, Xr, Xs, ee_b, epCT[0], Er, Es, epb1[0], hbuf);
    } else {            // edge input = stored edge_effect
      k_edge1<<<blocks1d(ewaves * 32, TPB), TPB, 0, stream>>>(
          ebuf, epCT[1], nullptr, nullptr, nullptr, nullptr, Er, Es, epb1[1], hbuf);
    }
    long e2waves = (long)BBATCH * NN * (HDIM >> 6);   // 512 waves
    k_edge2<<<blocks1d(e2waves * 32, TPB), TPB, 0, stream>>>(
        hbuf, epW2T[t], epb2[t], ebuf, agg);
    k_cvt<<<blocks1d(NRows * HDIM, TPB), TPB, 0, stream>>>(agg, aggB,
                                                           (int)(NRows * HDIM));
    // node update: g = relu(enc@A + eff@B + agg@C + b1); eff' = relu(g@W2 + b2)
    gemm(node_enc, npAT, neff, npBT, aggB, npCT, np_b1,
         NRows, HDIM, HDIM, 1, nullptr, gbuf);
    gemm(gbuf, npW2T, 0,0, 0,0, np_b2, NRows, HDIM, HDIM, 1, nullptr, node_eff);
  }

  // --- predictor: p = relu(edge_eff@prA + edge_enc(virtual)@prB + b1) ---
  k_edge1<<<blocks1d(ewaves * 32, TPB), TPB, 0, stream>>>(
      ebuf, prAT, Xr, Xs, ee_b, prBT, nullptr, nullptr, pr_b1, hbuf);
  // out = p @ pr_W2 + pr_b2   (M=131072, K=256, N=64, f32 output)
  gemm(hbuf, prW2T, 0,0, 0,0, pr_b2, ERows, HDIM, 64, 0, (float*)d_out, nullptr);
}
